// MultiHeadAttentionLayer_30107720745808
// MI455X (gfx1250) — matrix operand order repaired and checked by reading, not compiled, not run on hardware
//
#include <hip/hip_runtime.h>

// ---------------------------------------------------------------------------
// MI455X (gfx1250) multi-head attention: bf16 WMMA everywhere, fp32 softmax.
// B=2, S=2048, H=1024, NH=16, HD=64.  SCALE=8 -> logits = qk/64 + (1-m)*(-1e10)/8
// ---------------------------------------------------------------------------

#define Bn 2
#define Sn 2048
#define Hn 1024
#define NH 16
#define HD 64

typedef __attribute__((ext_vector_type(16))) __bf16 v16bf;
typedef __attribute__((ext_vector_type(8)))  __bf16 v8bf;    // 16 bytes
typedef __attribute__((ext_vector_type(8)))  float  v8f;

union Frag16 {
  v16bf v;
  v8bf  h[2];
};

__device__ __forceinline__ __bf16 f2bf(float f) {
  return (__bf16)f;                      // native v_cvt_*_bf16_f32 (RNE)
}

// Two 16-byte chunks -> one 16x bf16 fragment register set.
__device__ __forceinline__ v16bf load_frag(const __bf16* p0, const __bf16* p1) {
  Frag16 f;
  f.h[0] = *(const v8bf*)p0;
  f.h[1] = *(const v8bf*)p1;
  return f.v;
}

__device__ __forceinline__ v8f wmma_bf16(v16bf a, v16bf b, v8f c) {
  return __builtin_amdgcn_wmma_f32_16x16x32_bf16(
      false, a, false, b, (short)0, c, false, false);
}

// ---------------------------------------------------------------------------
// Tiled GEMM:  C[M=4096, N=1024] = X[M,1024] @ W[1024,1024] + bias
// Block tile 128(M) x 64(N), K-step 32.  256 threads = 8 waves (4x2),
// each wave computes a 32x32 patch as 2x2 WMMA tiles.
// Compile-time MODE: 0 -> bf16 out [b,h,s,d] (Q/K)
//                    2 -> bf16 out [b,h,d,s] (V transposed)
//                    3 -> fp32  out [m,n]    (final projection -> d_out)
// Compile-time XBF16: X is bf16 (attention output) instead of fp32.
// ---------------------------------------------------------------------------
template <int XBF16, int MODE>
__global__ __launch_bounds__(256)
void proj_gemm_kernel(const void* __restrict__ Xv,
                      const float* __restrict__ W,
                      const float* __restrict__ bias,
                      void* __restrict__ Out)
{
  __shared__ __bf16 Xs[128 * 40];   // [m][k], stride 40 (80B, 16B-aligned rows)
  __shared__ __bf16 Ws[64 * 40];    // transposed [n][k], stride 40

  const int tid  = threadIdx.x;
  const int lane = tid & 31;
  const int wave = tid >> 5;
  const int wm   = wave >> 1;       // 0..3  (M sub-block, 32 rows)
  const int wn   = wave & 1;        // 0..1  (N sub-block, 32 cols)
  const int half = lane >> 4;
  const int l16  = lane & 15;
  const int bm   = blockIdx.y * 128;
  const int bn   = blockIdx.x * 64;

  v8f acc[2][2] = {};

  for (int kt = 0; kt < Hn; kt += 32) {
    // ---- stage X tile (128 x 32) as bf16 (branch-free, templated) ----
#pragma unroll
    for (int it = 0; it < 4; ++it) {
      int idx = (it * 256 + tid) * 4;       // 4 contiguous elements
      int r = idx >> 5;
      int c = idx & 31;
      __bf16 b0, b1, b2, b3;
      if (XBF16) {
        const __bf16* xp = (const __bf16*)Xv + (size_t)(bm + r) * Hn + kt + c;
        b0 = xp[0]; b1 = xp[1]; b2 = xp[2]; b3 = xp[3];
      } else {
        const float* xp = (const float*)Xv + (size_t)(bm + r) * Hn + kt + c;
        b0 = f2bf(xp[0]); b1 = f2bf(xp[1]); b2 = f2bf(xp[2]); b3 = f2bf(xp[3]);
      }
      __bf16* d = &Xs[r * 40 + c];
      d[0] = b0; d[1] = b1; d[2] = b2; d[3] = b3;
    }
    // ---- stage W tile (32 x 64) transposed -> Ws[n][k] ----
#pragma unroll
    for (int it = 0; it < 2; ++it) {
      int idx = (it * 256 + tid) * 4;
      int r = idx >> 6;                     // k row 0..31
      int c = idx & 63;                     // n col
      const float* wp = W + (size_t)(kt + r) * Hn + bn + c;
#pragma unroll
      for (int i = 0; i < 4; ++i)
        Ws[(c + i) * 40 + r] = f2bf(wp[i]);
    }
    __syncthreads();

    // ---- fragments + 4 WMMAs ----
    v16bf af[2], bfr[2];
    {
      int kb = half * 16;
#pragma unroll
      for (int s = 0; s < 2; ++s) {
        const __bf16* p = &Xs[(wm * 32 + s * 16 + l16) * 40 + kb];
        af[s] = load_frag(p, p + 8);
      }
    }
    {
      int kb = half * 16;
#pragma unroll
      for (int s = 0; s < 2; ++s) {
        const __bf16* p = &Ws[(wn * 32 + s * 16 + l16) * 40 + kb];
        bfr[s] = load_frag(p, p + 8);
      }
    }
#pragma unroll
    for (int i = 0; i < 2; ++i)
#pragma unroll
      for (int j = 0; j < 2; ++j)
        acc[i][j] = wmma_bf16(af[i], bfr[j], acc[i][j]);

    __syncthreads();
  }

  // ---- epilogue: C layout row = r + 8*half, col = l16 ----
#pragma unroll
  for (int i = 0; i < 2; ++i) {
#pragma unroll
    for (int j = 0; j < 2; ++j) {
#pragma unroll
      for (int r = 0; r < 8; ++r) {
        int m = bm + wm * 32 + i * 16 + half * 8 + r;
        int n = bn + wn * 32 + j * 16 + l16;
        float v = acc[i][j][r] + bias[n];
        if (MODE == 3) {
          ((float*)Out)[(size_t)m * Hn + n] = v;
        } else {
          int b = m >> 11, s = m & (Sn - 1);
          int h = n >> 6,  d = n & (HD - 1);
          if (MODE == 2) {   // V transposed: [b][h][d][s]
            ((__bf16*)Out)[(((size_t)(b * NH + h) * HD + d) << 11) + s] = f2bf(v);
          } else {           // Q / K: [b][h][s][d]
            ((__bf16*)Out)[((((size_t)(b * NH + h) << 11) + s) << 6) + d] = f2bf(v);
          }
        }
      }
    }
  }
}

// ---------------------------------------------------------------------------
// Flash attention: one block = (b,h) x 64 query rows; 4 waves x 16 rows.
// S = Q K^T (WMMA), online softmax in fp32, P V (WMMA via per-wave LDS relayout).
// ---------------------------------------------------------------------------
__global__ __launch_bounds__(128)
void attn_kernel(const __bf16* __restrict__ Qb,   // [b,h,s,d] bf16
                 const __bf16* __restrict__ Kb,   // [b,h,s,d] bf16
                 const __bf16* __restrict__ Vt,   // [b,h,d,s] bf16
                 const float* __restrict__ mask,  // [B,1,1,S]
                 __bf16* __restrict__ Outb)       // [b,s,h*64+d] bf16
{
  __shared__ __bf16 Ps[4][16 * 72];   // per-wave P tile, stride 72 (144B rows)

  const int tid  = threadIdx.x;
  const int lane = tid & 31;
  const int wave = tid >> 5;
  const int half = lane >> 4;
  const int l16  = lane & 15;
  const int bh   = blockIdx.y;                // b*16 + h
  const int b    = bh >> 4;
  const int h    = bh & 15;
  const int q0   = blockIdx.x * 64 + wave * 16;

  const size_t headBase = (size_t)bh * Sn * HD;
  const __bf16* Qh = Qb + headBase;
  const __bf16* Kh = Kb + headBase;
  const __bf16* Vh = Vt + headBase;           // [d][s]

  // Q fragments: lane = q row, contiguous d.
  v16bf aq[2];
  {
    const __bf16* qp = Qh + (size_t)(q0 + l16) * HD;
    int kb = half * 16;
    aq[0] = load_frag(qp + kb,      qp + kb + 8);
    aq[1] = load_frag(qp + 32 + kb, qp + 32 + kb + 8);
  }

  float m_r[8], l_r[8];
  v8f o[4] = {};
#pragma unroll
  for (int r = 0; r < 8; ++r) { m_r[r] = -3.0e38f; l_r[r] = 0.0f; }

  const float inv64 = 1.0f / 64.0f;           // 1/SCALE^2
  __bf16* psw = Ps[wave];

  for (int kt = 0; kt < Sn; kt += 64) {
    // Prefetch next K/V tiles into cache (global_prefetch_b8).
    if (kt + 64 < Sn) {
      __builtin_prefetch(Kh + (size_t)(kt + 64 + lane) * HD, 0, 1);
      __builtin_prefetch(Vh + (size_t)lane * Sn * 2 + kt + 64, 0, 1);
    }

    // ---- S = Q K^T : 4 key sub-tiles x 2 d-chunks ----
    v8f s[4] = {};
#pragma unroll
    for (int n = 0; n < 4; ++n) {
      const __bf16* kp = Kh + (size_t)(kt + n * 16 + l16) * HD;
      int kb = half * 16;
      v16bf b0 = load_frag(kp + kb,      kp + kb + 8);
      v16bf b1 = load_frag(kp + 32 + kb, kp + 32 + kb + 8);
      s[n] = wmma_bf16(aq[0], b0, s[n]);
      s[n] = wmma_bf16(aq[1], b1, s[n]);
    }

    // ---- logits + online softmax (fp32) ----
    float mk[4];
#pragma unroll
    for (int n = 0; n < 4; ++n)
      mk[n] = (1.0f - mask[(size_t)b * Sn + kt + n * 16 + l16]) * (-1.25e9f);

    float lo[4][8];
    float mnew[8];
#pragma unroll
    for (int r = 0; r < 8; ++r) mnew[r] = m_r[r];
#pragma unroll
    for (int n = 0; n < 4; ++n)
#pragma unroll
      for (int r = 0; r < 8; ++r) {
        lo[n][r] = s[n][r] * inv64 + mk[n];
        mnew[r]  = fmaxf(mnew[r], lo[n][r]);
      }
#pragma unroll
    for (int r = 0; r < 8; ++r) {
      mnew[r] = fmaxf(mnew[r], __shfl_xor(mnew[r], 1, 32));
      mnew[r] = fmaxf(mnew[r], __shfl_xor(mnew[r], 2, 32));
      mnew[r] = fmaxf(mnew[r], __shfl_xor(mnew[r], 4, 32));
      mnew[r] = fmaxf(mnew[r], __shfl_xor(mnew[r], 8, 32));
    }

    float alpha[8], sum[8];
#pragma unroll
    for (int r = 0; r < 8; ++r) {
      alpha[r] = __expf(m_r[r] - mnew[r]);
      sum[r]   = 0.0f;
    }
#pragma unroll
    for (int n = 0; n < 4; ++n)
#pragma unroll
      for (int r = 0; r < 8; ++r) {
        float p = __expf(lo[n][r] - mnew[r]);
        sum[r] += p;
        psw[(half * 8 + r) * 72 + n * 16 + l16] = f2bf(p);
      }
#pragma unroll
    for (int r = 0; r < 8; ++r) {
      sum[r] += __shfl_xor(sum[r], 1, 32);
      sum[r] += __shfl_xor(sum[r], 2, 32);
      sum[r] += __shfl_xor(sum[r], 4, 32);
      sum[r] += __shfl_xor(sum[r], 8, 32);
      l_r[r] = l_r[r] * alpha[r] + sum[r];
      m_r[r] = mnew[r];
    }
#pragma unroll
    for (int t = 0; t < 4; ++t)
#pragma unroll
      for (int r = 0; r < 8; ++r)
        o[t][r] *= alpha[r];

    // ---- O += P V : reload P as A-fragments (same wave, LDS in-order) ----
    v16bf ap[2];
    {
      const __bf16* pp = &psw[l16 * 72];
      int kb = half * 16;
      ap[0] = load_frag(pp + kb,      pp + kb + 8);
      ap[1] = load_frag(pp + 32 + kb, pp + 32 + kb + 8);
    }
#pragma unroll
    for (int t = 0; t < 4; ++t) {
      const __bf16* vp = Vh + (size_t)(t * 16 + l16) * Sn + kt;
      int kb = half * 16;
      v16bf b0 = load_frag(vp + kb,      vp + kb + 8);
      v16bf b1 = load_frag(vp + 32 + kb, vp + 32 + kb + 8);
      o[t] = wmma_bf16(ap[0], b0, o[t]);
      o[t] = wmma_bf16(ap[1], b1, o[t]);
    }
  }

  // ---- normalize + store bf16 [b, q, h*64 + d] ----
#pragma unroll
  for (int r = 0; r < 8; ++r) {
    float inv = 1.0f / l_r[r];
    int q = q0 + half * 8 + r;
#pragma unroll
    for (int t = 0; t < 4; ++t) {
      int d = t * 16 + l16;
      Outb[((size_t)b * Sn + q) * Hn + h * HD + d] = f2bf(o[t][r] * inv);
    }
  }
}

// ---------------------------------------------------------------------------
// Launch: 3 projections -> attention -> output projection.
// Workspace: Q,K,Vt,Attn  (each 4096*1024 bf16 = 8 MiB, total 32 MiB).
// ---------------------------------------------------------------------------
extern "C" void kernel_launch(void* const* d_in, const int* in_sizes, int n_in,
                              void* d_out, int out_size, void* d_ws, size_t ws_size,
                              hipStream_t stream) {
  (void)in_sizes; (void)n_in; (void)out_size; (void)ws_size;

  const float* query = (const float*)d_in[0];
  const float* key   = (const float*)d_in[1];
  const float* value = (const float*)d_in[2];
  const float* amask = (const float*)d_in[3];
  const float* wq = (const float*)d_in[4];
  const float* bq = (const float*)d_in[5];
  const float* wk = (const float*)d_in[6];
  const float* bk = (const float*)d_in[7];
  const float* wv = (const float*)d_in[8];
  const float* bv = (const float*)d_in[9];
  const float* wo = (const float*)d_in[10];
  const float* bo = (const float*)d_in[11];

  const size_t nElem = (size_t)Bn * Sn * Hn;   // 4 Mi elements
  __bf16* Qb = (__bf16*)d_ws;
  __bf16* Kb = Qb + nElem;
  __bf16* Vt = Kb + nElem;
  __bf16* At = Vt + nElem;

  dim3 gp(Hn / 64, (Bn * Sn) / 128);   // (16, 32)
  dim3 bp(256);

  proj_gemm_kernel<0, 0><<<gp, bp, 0, stream>>>(query, wq, bq, Qb);
  proj_gemm_kernel<0, 0><<<gp, bp, 0, stream>>>(key,   wk, bk, Kb);
  proj_gemm_kernel<0, 2><<<gp, bp, 0, stream>>>(value, wv, bv, Vt);

  attn_kernel<<<dim3(Sn / 64, Bn * NH), 128, 0, stream>>>(Qb, Kb, Vt, amask, At);

  proj_gemm_kernel<1, 3><<<gp, bp, 0, stream>>>(At, wo, bo, d_out);
}
